// Seq2SeqLSTM_24515673325999
// MI455X (gfx1250) — matrix operand order repaired and checked by reading, not compiled, not run on hardware
//
#include <hip/hip_runtime.h>

// ---------------------------------------------------------------------------
// Seq2Seq LSTM + Bahdanau attention for MI455X (gfx1250, wave32, WMMA).
// B=16, S=128, T=65 (Td=64), D=256, enc hid=256/dir, dec hid=512, V=32000.
// Strategy: f16 weights + f32 accum via v_wmma_f32_16x16x32_f16.
//   - input-side LSTM matmuls hoisted to big GEMMs over the whole sequence
//   - recurrence done by persistent workgroups with h/c/gates in LDS
//   - vocab projection = 1024 x 32000 x 1024 f16 WMMA GEMM (dominant cost)
// Round-1 fix: branchless GEMM inner loop (no exec-mask predication/spills).
// Round-2 fix: do NOT unroll the encoder recurrence K-loop; the 16 hoisted
// loop-invariant Whh B-tiles (128 VGPRs) were spilling to scratch at
// 1024-thread occupancy. unroll(1) keeps only 2 B-tiles live; Whh re-reads
// hit L2 and hide behind 8 waves/SIMD.
// ---------------------------------------------------------------------------

typedef __attribute__((ext_vector_type(16))) _Float16 v16h;
typedef __attribute__((ext_vector_type(8)))  float    v8f;

// ---- WMMA wrapper ----------------------------------------------------------
__device__ __forceinline__ v8f wmma16(v16h a, v16h b, v8f c) {
  // emits v_wmma_f32_16x16x32_f16
  return __builtin_amdgcn_wmma_f32_16x16x32_f16(
      false, a, false, b, (short)0, c, false, false);
}

// ---- A tile loader: 16x32 f16, row-major source, contiguous K per lane half -----
// lanes 0-15: M=lane, VGPR0-7 K=0..15
// lanes 16-31: M=lane-16, VGPR0-7 K=16..31
__device__ __forceinline__ v16h load_A16(const _Float16* A, long lda) {
  const int lane = threadIdx.x & 31;
  const int hs = lane >> 4;
  const _Float16* row = A + (long)(lane & 15) * lda + 16 * hs;
  v16h a;
#pragma unroll
  for (int i = 0; i < 4; ++i) { a[2 * i] = row[2 * i]; a[2 * i + 1] = row[2 * i + 1]; }
  const _Float16* row2 = row + 8;
#pragma unroll
  for (int i = 0; i < 4; ++i) { a[8 + 2 * i] = row2[2 * i]; a[9 + 2 * i] = row2[2 * i + 1]; }
  return a;
}

// ---- B tile loader from W[N,K] row-major (B = W^T): contiguous K per lane --
// lanes 0-15: N=lane, VGPRj K=2j,2j+1 ; lanes 16-31: same N, K=16+2j,16+2j+1
__device__ __forceinline__ v16h load_B_nk(const _Float16* W, long ldw) {
  const int lane = threadIdx.x & 31;
  const _Float16* p = W + (long)(lane & 15) * ldw + ((lane >> 4) << 4);
  v16h b;
#pragma unroll
  for (int j = 0; j < 8; ++j) { b[2 * j] = p[2 * j]; b[2 * j + 1] = p[2 * j + 1]; }
  return b;
}

// ---- B tile loader from Bm[K,N] row-major (strided along K) ----------------
__device__ __forceinline__ v16h load_B_kn(const _Float16* Bm, long ldb) {
  const int lane = threadIdx.x & 31;
  const _Float16* col = Bm + (long)((lane >> 4) * 16) * ldb + (lane & 15);
  v16h b;
#pragma unroll
  for (int j = 0; j < 8; ++j) {
    b[2 * j]     = col[(long)(2 * j) * ldb];
    b[2 * j + 1] = col[(long)(2 * j + 1) * ldb];
  }
  return b;
}

__device__ __forceinline__ float sigf(float x) { return 1.f / (1.f + __expf(-x)); }

// ===========================================================================
// Generic WMMA GEMM: C[M,N](f32 and/or f16) = A[M,K](f16) * op(B) + bias
//   BNK=1 : B given as W[N,K] row-major (C = A * W^T)
//   BNK=0 : B given as Bm[K,N] row-major (C = A * Bm)
// block = 256 threads = 8 waves arranged 4(M)x2(N); wave tile = 64x32.
// REQUIREMENT: M % 64 == 0 and N % 64 == 0 (true for all call sites), so a
// wave is either fully valid or fully out of range -> branchless K loop.
// ===========================================================================
template <int BNK>
__global__ __launch_bounds__(256)
void gemm_wmma_kernel(const _Float16* __restrict__ A, long sAb, int lda,
                      const _Float16* __restrict__ Bm, long sBb, int ldb,
                      const float* __restrict__ bias,
                      float* __restrict__ C32, _Float16* __restrict__ C16,
                      long sCb, int ldc, int M, int N, int K) {
  const int wave = threadIdx.x >> 5;
  const long m0 = (long)blockIdx.y * 256 + (wave >> 1) * 64;
  const long n0 = (long)blockIdx.x * 64 + (wave & 1) * 32;
  if (m0 >= M) return;  // whole-wave exit (M % 64 == 0); no barriers here

  A  += (long)blockIdx.z * sAb + m0 * (long)lda;
  Bm += (long)blockIdx.z * sBb;
  const long coff = (long)blockIdx.z * sCb;

  const v8f z = {};
  v8f acc[4][2];
#pragma unroll
  for (int i = 0; i < 4; ++i) { acc[i][0] = z; acc[i][1] = z; }

  for (int k0 = 0; k0 < K; k0 += 32) {
    v16h bt0, bt1;
    if (BNK) {
      bt0 = load_B_nk(Bm + n0 * (long)ldb + k0, ldb);
      bt1 = load_B_nk(Bm + (n0 + 16) * (long)ldb + k0, ldb);
    } else {
      bt0 = load_B_kn(Bm + (long)k0 * ldb + n0, ldb);
      bt1 = load_B_kn(Bm + (long)k0 * ldb + (n0 + 16), ldb);
    }
#pragma unroll
    for (int i = 0; i < 4; ++i) {
      v16h at = load_A16(A + (long)(16 * i) * lda + k0, lda);
      acc[i][0] = wmma16(at, bt0, acc[i][0]);
      acc[i][1] = wmma16(at, bt1, acc[i][1]);
    }
  }

  const int lane = threadIdx.x & 31;
  const int cn = lane & 15, chs = lane >> 4;
  float bsv0 = 0.f, bsv1 = 0.f;
  if (bias) { bsv0 = bias[n0 + cn]; bsv1 = bias[n0 + 16 + cn]; }
#pragma unroll
  for (int i = 0; i < 4; ++i) {
#pragma unroll
    for (int j = 0; j < 2; ++j) {
      float bv = j ? bsv1 : bsv0;
#pragma unroll
      for (int v = 0; v < 8; ++v) {
        long m = m0 + 16 * i + v + 8 * chs;
        long n = n0 + 16 * j + cn;
        float val = acc[i][j][v] + bv;
        if (C32) C32[coff + m * (long)ldc + n] = val;
        if (C16) C16[coff + m * (long)ldc + n] = (_Float16)val;
      }
    }
  }
}

// ===========================================================================
// Encoder LSTM layer: one workgroup per direction (grid.x=2), 1024 thr/32 wv.
// Per step: gates[16,1024] = Xpre[:,s,:] + h @ Whh^T via WMMA, then cell.
// LDS: gates 64KB f32, c 16KB f32, h 8KB f16.
// K-loop deliberately NOT unrolled (see header comment).
// ===========================================================================
__global__ void lstm_enc_kernel(const float* __restrict__ XpreF,
                                const float* __restrict__ XpreB,
                                const _Float16* __restrict__ WhhF,
                                const _Float16* __restrict__ WhhB,
                                _Float16* __restrict__ outSeq,   // [2048,512]
                                float* __restrict__ hsOut,       // [16,512]
                                float* __restrict__ csOut) {     // [16,512]
  const int dir = blockIdx.x;  // 0=fwd, 1=bwd
  const float* Xpre = dir ? XpreB : XpreF;
  const _Float16* Whh = dir ? WhhB : WhhF;
  extern __shared__ char smemRaw[];
  float* gates = (float*)smemRaw;                   // [16][1024]
  float* cst   = gates + 16 * 1024;                 // [16][256]
  _Float16* hh = (_Float16*)(cst + 16 * 256);       // [16][256]
  const int tid = threadIdx.x;
  for (int i = tid; i < 16 * 256; i += 1024) { cst[i] = 0.f; hh[i] = (_Float16)0.f; }
  __syncthreads();

  const int wave = tid >> 5, lane = tid & 31;
  const int n0 = wave * 32;
  const int cn = lane & 15, chs = lane >> 4;
  const _Float16* hhG = hh;
  const _Float16* Wp0 = Whh + (long)n0 * 256;         // this wave's 32-col slice
  const _Float16* Wp1 = Whh + (long)(n0 + 16) * 256;

#pragma unroll 1
  for (int t = 0; t < 128; ++t) {
    const int sIn = dir ? (127 - t) : t;
    v8f acc0 = {}, acc1 = {};
#pragma unroll 1
    for (int k0 = 0; k0 < 256; k0 += 32) {
      v16h a = load_A16(hhG + k0, 256);
      acc0 = wmma16(a, load_B_nk(Wp0 + k0, 256), acc0);
      acc1 = wmma16(a, load_B_nk(Wp1 + k0, 256), acc1);
    }
#pragma unroll
    for (int v = 0; v < 8; ++v) {
      int b = v + 8 * chs;
      long xrow = ((long)b * 128 + sIn) * 1024;
      int n = n0 + cn;
      gates[b * 1024 + n]      = acc0[v] + Xpre[xrow + n];
      gates[b * 1024 + n + 16] = acc1[v] + Xpre[xrow + n + 16];
    }
    __syncthreads();
#pragma unroll
    for (int p = 0; p < 4; ++p) {
      int idx = tid + p * 1024;            // (b,j) over 16x256
      int b = idx >> 8, j = idx & 255;
      float gi = gates[b * 1024 + j];
      float gf = gates[b * 1024 + 256 + j];
      float gg = gates[b * 1024 + 512 + j];
      float go = gates[b * 1024 + 768 + j];
      float c = sigf(gf) * cst[idx] + sigf(gi) * tanhf(gg);
      float h = sigf(go) * tanhf(c);
      cst[idx] = c;
      hh[idx] = (_Float16)h;
      outSeq[((long)b * 128 + sIn) * 512 + dir * 256 + j] = (_Float16)h;
      if (t == 127) {
        hsOut[b * 512 + dir * 256 + j] = h;
        csOut[b * 512 + dir * 256 + j] = c;
      }
    }
    __syncthreads();
  }
}

// ===========================================================================
// Decoder LSTM layer: hidden 512, gates 2048, 64 steps, one workgroup.
// LDS: gates 128KB f32, c 32KB f32, h 16KB f16 (176KB < 320KB/WGP).
// ===========================================================================
__global__ void lstm_dec_kernel(const float* __restrict__ Xpre,    // [1024,2048]
                                const _Float16* __restrict__ Whh,  // [2048,512]
                                const float* __restrict__ h0,      // [16,512]
                                const float* __restrict__ c0,      // [16,512]
                                _Float16* __restrict__ outSeq) {   // [1024,512]
  extern __shared__ char smemRaw[];
  float* gates = (float*)smemRaw;                   // [16][2048]
  float* cst   = gates + 16 * 2048;                 // [16][512]
  _Float16* hh = (_Float16*)(cst + 16 * 512);       // [16][512]
  const int tid = threadIdx.x;
  for (int i = tid; i < 16 * 512; i += 1024) { cst[i] = c0[i]; hh[i] = (_Float16)h0[i]; }
  __syncthreads();

  const int wave = tid >> 5, lane = tid & 31;
  const int n0 = wave * 64;
  const int cn = lane & 15, chs = lane >> 4;
  const _Float16* hhG = hh;

#pragma unroll 1
  for (int t = 0; t < 64; ++t) {
    const v8f z = {};
    v8f acc[4]; acc[0] = z; acc[1] = z; acc[2] = z; acc[3] = z;
#pragma unroll 1
    for (int k0 = 0; k0 < 512; k0 += 32) {
      v16h a = load_A16(hhG + k0, 512);
#pragma unroll
      for (int j = 0; j < 4; ++j)
        acc[j] = wmma16(a, load_B_nk(Whh + (long)(n0 + 16 * j) * 512 + k0, 512), acc[j]);
    }
#pragma unroll
    for (int j = 0; j < 4; ++j)
#pragma unroll
      for (int v = 0; v < 8; ++v) {
        int b = v + 8 * chs;
        int n = n0 + 16 * j + cn;
        gates[b * 2048 + n] = acc[j][v] + Xpre[((long)b * 64 + t) * 2048 + n];
      }
    __syncthreads();
#pragma unroll
    for (int p = 0; p < 8; ++p) {
      int idx = tid + p * 1024;            // (b,j) over 16x512
      int b = idx >> 9, j = idx & 511;
      float gi = gates[b * 2048 + j];
      float gf = gates[b * 2048 + 512 + j];
      float gg = gates[b * 2048 + 1024 + j];
      float go = gates[b * 2048 + 1536 + j];
      float c = sigf(gf) * cst[idx] + sigf(gi) * tanhf(gg);
      float h = sigf(go) * tanhf(c);
      cst[idx] = c;
      hh[idx] = (_Float16)h;
      outSeq[((long)b * 64 + t) * 512 + j] = (_Float16)h;
    }
    __syncthreads();
  }
}

// ===========================================================================
// Attention: energy, softmax
// ===========================================================================
__global__ void energy_kernel(const float* __restrict__ encP,   // [2048,512]
                              const float* __restrict__ decP,   // [1024,512]
                              const float* __restrict__ vvec,   // [512]
                              float* __restrict__ energy) {     // [16,64,128]
  long gwave = ((long)blockIdx.x * blockDim.x + threadIdx.x) >> 5;
  int lane = threadIdx.x & 31;
  int s = (int)(gwave & 127);
  int t = (int)((gwave >> 7) & 63);
  int b = (int)(gwave >> 13);
  const float* ep = encP + ((long)b * 128 + s) * 512;
  const float* dp = decP + ((long)b * 64 + t) * 512;
  float sum = 0.f;
#pragma unroll
  for (int i = 0; i < 16; ++i) {
    int d = lane + 32 * i;
    sum += vvec[d] * tanhf(ep[d] + dp[d]);
  }
  for (int off = 16; off > 0; off >>= 1) sum += __shfl_xor(sum, off, 32);
  if (lane == 0) energy[gwave] = sum;
}

__global__ void softmax_kernel(const float* __restrict__ energy,  // [1024,128]
                               _Float16* __restrict__ attn) {     // [1024,128] f16
  int row = (int)(((long)blockIdx.x * blockDim.x + threadIdx.x) >> 5);
  int lane = threadIdx.x & 31;
  const float* e = energy + (long)row * 128;
  float vals[4]; float mx = -3.4e38f;
#pragma unroll
  for (int i = 0; i < 4; ++i) { vals[i] = e[lane + 32 * i]; mx = fmaxf(mx, vals[i]); }
  for (int off = 16; off > 0; off >>= 1) mx = fmaxf(mx, __shfl_xor(mx, off, 32));
  float sum = 0.f;
#pragma unroll
  for (int i = 0; i < 4; ++i) { vals[i] = __expf(vals[i] - mx); sum += vals[i]; }
  for (int off = 16; off > 0; off >>= 1) sum += __shfl_xor(sum, off, 32);
  float inv = 1.f / sum;
  _Float16* a = attn + (long)row * 128;
#pragma unroll
  for (int i = 0; i < 4; ++i) a[lane + 32 * i] = (_Float16)(vals[i] * inv);
}

// ===========================================================================
// Small helper kernels
// ===========================================================================
__global__ void cvt_f16_kernel(const float* __restrict__ src, _Float16* __restrict__ dst, long n) {
  long i = (long)blockIdx.x * blockDim.x + threadIdx.x;
  long stride = (long)gridDim.x * blockDim.x;
  for (; i < n; i += stride) dst[i] = (_Float16)src[i];
}

__global__ void addbias_kernel(const float* __restrict__ a, const float* __restrict__ b,
                               float* __restrict__ dst, int n) {
  int i = blockIdx.x * blockDim.x + threadIdx.x;
  if (i < n) dst[i] = a[i] + b[i];
}

__global__ void embed_src_kernel(const int* __restrict__ x, const float* __restrict__ emb,
                                 _Float16* __restrict__ out) {  // [2048,256]
  long i = (long)blockIdx.x * blockDim.x + threadIdx.x;
  if (i >= 2048L * 256) return;
  long row = i >> 8; int d = (int)(i & 255);
  out[i] = (_Float16)emb[(long)x[row] * 256 + d];
}

__global__ void embed_tgt_kernel(const int* __restrict__ y, const float* __restrict__ emb,
                                 _Float16* __restrict__ out) {  // [1024,1024], top half zero
  long i = (long)blockIdx.x * blockDim.x + threadIdx.x;
  if (i >= 1024L * 1024) return;
  long r = i >> 10; int c = (int)(i & 1023);
  int b = (int)(r >> 6), t = (int)(r & 63);
  _Float16 v = (_Float16)0.f;
  if (c < 512) v = (_Float16)emb[(long)y[b * 65 + t] * 512 + c];
  out[i] = v;
}

__global__ void pack_kernel(const _Float16* __restrict__ dec,   // [1024,512]
                            _Float16* __restrict__ comb) {      // [1024,1024] cols 0-511
  long i = (long)blockIdx.x * blockDim.x + threadIdx.x;
  if (i >= 1024L * 512) return;
  long r = i >> 9; int c = (int)(i & 511);
  comb[r * 1024 + c] = dec[i];
}

// ===========================================================================
// Host orchestration
// ===========================================================================
extern "C" void kernel_launch(void* const* d_in, const int* in_sizes, int n_in,
                              void* d_out, int out_size, void* d_ws, size_t ws_size,
                              hipStream_t stream) {
  (void)in_sizes; (void)n_in; (void)out_size; (void)ws_size;

  // ---- input pointers (JAX pytree sorted-key flatten order) ----
  const float* W_h   = (const float*)d_in[0];   // [512,512]
  const float* W_out = (const float*)d_in[1];   // [32000,1024]
  const float* W_s   = (const float*)d_in[2];   // [512,512]
  const float* b_out = (const float*)d_in[3];   // [32000]
  const float *decWhh[3], *decWih[3], *decBhh[3], *decBih[3];
  for (int l = 0; l < 3; ++l) {
    decWhh[l] = (const float*)d_in[4 + 4 * l];
    decWih[l] = (const float*)d_in[5 + 4 * l];
    decBhh[l] = (const float*)d_in[6 + 4 * l];
    decBih[l] = (const float*)d_in[7 + 4 * l];
  }
  // enc: per layer {bwd, fwd}, each {Whh, Wih, bhh, bih}; dir 0=fwd, 1=bwd
  const float *encWhh[3][2], *encWih[3][2], *encBhh[3][2], *encBih[3][2];
  for (int l = 0; l < 3; ++l) {
    encWhh[l][1] = (const float*)d_in[16 + 8 * l];
    encWih[l][1] = (const float*)d_in[17 + 8 * l];
    encBhh[l][1] = (const float*)d_in[18 + 8 * l];
    encBih[l][1] = (const float*)d_in[19 + 8 * l];
    encWhh[l][0] = (const float*)d_in[20 + 8 * l];
    encWih[l][0] = (const float*)d_in[21 + 8 * l];
    encBhh[l][0] = (const float*)d_in[22 + 8 * l];
    encBih[l][0] = (const float*)d_in[23 + 8 * l];
  }
  const float* src_emb = (const float*)d_in[40];
  const float* tgt_emb = (const float*)d_in[41];
  const float* vvec    = (const float*)d_in[42];
  const int*   x       = (const int*)d_in[43];
  const int*   y       = (const int*)d_in[44];

  // ---- workspace bump allocator (deterministic layout) ----
  size_t off = 0;
  auto alloc = [&](size_t bytes) -> char* {
    char* p = (char*)d_ws + off;
    off += (bytes + 255) & ~(size_t)255;
    return p;
  };
  _Float16* wWh   = (_Float16*)alloc(512L * 512 * 2);
  _Float16* wWs   = (_Float16*)alloc(512L * 512 * 2);
  _Float16* wWout = (_Float16*)alloc(32000L * 1024 * 2);
  _Float16 *wDecWhh[3], *wDecWih[3]; float* bDec[3];
  for (int l = 0; l < 3; ++l) {
    wDecWhh[l] = (_Float16*)alloc(2048L * 512 * 2);
    wDecWih[l] = (_Float16*)alloc(2048L * (l == 0 ? 1024 : 512) * 2);
    bDec[l]    = (float*)alloc(2048L * 4);
  }
  _Float16 *wEncWhh[3][2], *wEncWih[3][2]; float* bEnc[3][2];
  for (int l = 0; l < 3; ++l)
    for (int d = 0; d < 2; ++d) {
      wEncWhh[l][d] = (_Float16*)alloc(1024L * 256 * 2);
      wEncWih[l][d] = (_Float16*)alloc(1024L * (l == 0 ? 256 : 512) * 2);
      bEnc[l][d]    = (float*)alloc(1024L * 4);
    }
  _Float16* embS   = (_Float16*)alloc(2048L * 256 * 2);
  float* XpreF     = (float*)alloc(2048L * 1024 * 4);
  float* XpreB     = (float*)alloc(2048L * 1024 * 4);
  _Float16* encSeq = (_Float16*)alloc(2048L * 512 * 2);
  float* hsBuf     = (float*)alloc(3L * 16 * 512 * 4);
  float* csBuf     = (float*)alloc(3L * 16 * 512 * 4);
  _Float16* decIn  = (_Float16*)alloc(1024L * 1024 * 2);
  float* XpreD     = (float*)alloc(1024L * 2048 * 4);
  _Float16* decSeq = (_Float16*)alloc(1024L * 512 * 2);
  float* encProj   = (float*)alloc(2048L * 512 * 4);
  float* decProj   = (float*)alloc(1024L * 512 * 4);
  float* energyB   = (float*)alloc(16L * 64 * 128 * 4);
  _Float16* attnB  = (_Float16*)alloc(16L * 64 * 128 * 2);
  _Float16* comb   = (_Float16*)alloc(1024L * 1024 * 2);

  auto cvt = [&](const float* s, _Float16* d, long n) {
    long blocks = (n + 255) / 256; if (blocks > 8192) blocks = 8192;
    cvt_f16_kernel<<<dim3((unsigned)blocks), dim3(256), 0, stream>>>(s, d, n);
  };
  auto addb = [&](const float* a, const float* b, float* d, int n) {
    addbias_kernel<<<dim3((n + 255) / 256), dim3(256), 0, stream>>>(a, b, d, n);
  };
  // C = A * W^T (+bias): W given as [N,K] row-major
  auto gemmNK = [&](const _Float16* A, long sAb, int lda,
                    const _Float16* W, long sWb, int ldw,
                    const float* bias, float* C32, _Float16* C16, long sCb, int ldc,
                    int M, int N, int K, int batch) {
    dim3 g((unsigned)(N / 64), (unsigned)((M + 255) / 256), (unsigned)batch);
    gemm_wmma_kernel<1><<<g, dim3(256), 0, stream>>>(A, sAb, lda, W, sWb, ldw,
                                                     bias, C32, C16, sCb, ldc, M, N, K);
  };
  // C = A * Bm (+bias): Bm given as [K,N] row-major
  auto gemmKN = [&](const _Float16* A, long sAb, int lda,
                    const _Float16* Bm, long sBb, int ldb,
                    const float* bias, float* C32, _Float16* C16, long sCb, int ldc,
                    int M, int N, int K, int batch) {
    dim3 g((unsigned)(N / 64), (unsigned)((M + 255) / 256), (unsigned)batch);
    gemm_wmma_kernel<0><<<g, dim3(256), 0, stream>>>(A, sAb, lda, Bm, sBb, ldb,
                                                     bias, C32, C16, sCb, ldc, M, N, K);
  };

  // ---- weight conversions (f32 -> f16) and bias combines ----
  cvt(W_h, wWh, 512L * 512);
  cvt(W_s, wWs, 512L * 512);
  cvt(W_out, wWout, 32000L * 1024);
  for (int l = 0; l < 3; ++l) {
    cvt(decWhh[l], wDecWhh[l], 2048L * 512);
    cvt(decWih[l], wDecWih[l], 2048L * (l == 0 ? 1024 : 512));
    addb(decBih[l], decBhh[l], bDec[l], 2048);
    for (int d = 0; d < 2; ++d) {
      cvt(encWhh[l][d], wEncWhh[l][d], 1024L * 256);
      cvt(encWih[l][d], wEncWih[l][d], 1024L * (l == 0 ? 256 : 512));
      addb(encBih[l][d], encBhh[l][d], bEnc[l][d], 1024);
    }
  }

  // ---- encoder ----
  embed_src_kernel<<<dim3(2048), dim3(256), 0, stream>>>(x, src_emb, embS);
  const size_t ENC_SMEM = 16 * 1024 * 4 + 16 * 256 * 4 + 16 * 256 * 2;   // 90112
  const _Float16* encInp = embS; int encK = 256;
  for (int l = 0; l < 3; ++l) {
    gemmNK(encInp, 0, encK, wEncWih[l][0], 0, encK, bEnc[l][0],
           XpreF, nullptr, 0, 1024, 2048, 1024, encK, 1);
    gemmNK(encInp, 0, encK, wEncWih[l][1], 0, encK, bEnc[l][1],
           XpreB, nullptr, 0, 1024, 2048, 1024, encK, 1);
    lstm_enc_kernel<<<dim3(2), dim3(1024), ENC_SMEM, stream>>>(
        XpreF, XpreB, wEncWhh[l][0], wEncWhh[l][1], encSeq,
        hsBuf + (long)l * 16 * 512, csBuf + (long)l * 16 * 512);
    encInp = encSeq; encK = 512;
  }

  // ---- decoder ----
  embed_tgt_kernel<<<dim3(4096), dim3(256), 0, stream>>>(y, tgt_emb, decIn);
  const size_t DEC_SMEM = 16 * 2048 * 4 + 16 * 512 * 4 + 16 * 512 * 2;   // 180224
  const _Float16* decInp = decIn; int decK = 1024;
  for (int l = 0; l < 3; ++l) {
    gemmNK(decInp, 0, decK, wDecWih[l], 0, decK, bDec[l],
           XpreD, nullptr, 0, 2048, 1024, 2048, decK, 1);
    lstm_dec_kernel<<<dim3(1), dim3(1024), DEC_SMEM, stream>>>(
        XpreD, wDecWhh[l], hsBuf + (long)l * 16 * 512, csBuf + (long)l * 16 * 512, decSeq);
    decInp = decSeq; decK = 512;
  }

  // ---- attention ----
  gemmNK(encSeq, 0, 512, wWh, 0, 512, nullptr, encProj, nullptr, 0, 512, 2048, 512, 512, 1);
  gemmNK(decSeq, 0, 512, wWs, 0, 512, nullptr, decProj, nullptr, 0, 512, 1024, 512, 512, 1);
  energy_kernel<<<dim3(16384), dim3(256), 0, stream>>>(encProj, decProj, vvec, energyB);
  softmax_kernel<<<dim3(128), dim3(256), 0, stream>>>(energyB, attnB);
  // ctx[b] = attn[b] (64x128) @ enc_out[b] (128x512) -> f16 into comb cols 512..1023
  gemmKN(attnB, 64L * 128, 128, encSeq, 128L * 512, 512, nullptr,
         nullptr, comb + 512, 64L * 1024, 1024, 64, 512, 128, 16);
  pack_kernel<<<dim3(2048), dim3(256), 0, stream>>>(decSeq, comb);

  // ---- vocab projection: [1024,1024] x [32000,1024]^T + b_out -> d_out ----
  gemmNK(comb, 0, 1024, wWout, 0, 1024, b_out,
         (float*)d_out, nullptr, 0, 32000, 1024, 32000, 1024, 1);
}
